// Codebook_84653805404166
// MI455X (gfx1250) — matrix operand order repaired and checked by reading, not compiled, not run on hardware
//
#include <hip/hip_runtime.h>
#include <hip/hip_bf16.h>

// ---------------------------------------------------------------------------
// VQ codebook quantization for MI455X (gfx1250, wave32, WMMA).
//   dist(m,n) = ||z_m||^2 - 2 z_m.w_n + ||w_n||^2 ; argmin_n drops ||z_m||^2.
//   GEMM on bf16 WMMA (f32 accum): v_wmma_f32_16x16x32_bf16.
//   Codebook slab staged memory->LDS with GLOBAL_LOAD_ASYNC_TO_LDS_B128.
// ---------------------------------------------------------------------------

typedef __bf16  bf16x16 __attribute__((ext_vector_type(16)));
typedef __bf16  bf16x8  __attribute__((ext_vector_type(8)));
typedef __bf16  bf16x4  __attribute__((ext_vector_type(4)));
typedef unsigned short u16x4 __attribute__((ext_vector_type(4)));
typedef float   f32x8   __attribute__((ext_vector_type(8)));
typedef float   f32x4   __attribute__((ext_vector_type(4)));

#define KCODES 1024
#define DDIM   512
#define MROWS  32768           // B*H*W = 32*32*32
#define MTILE  128             // rows per workgroup (8 waves x 16 rows)
#define LDA    520             // padded bf16 row stride (bank-conflict break, 16B align)
#define LDB    520

__device__ __forceinline__ unsigned short f2bf(float f) {
  // round-to-nearest-even f32 -> bf16
  unsigned u = __float_as_uint(f);
  u += 0x7FFFu + ((u >> 16) & 1u);
  return (unsigned short)(u >> 16);
}

// ---------------------------------------------------------------------------
// Kernel 1: codebook f32 -> bf16 (ws) + per-code squared norms.
// ---------------------------------------------------------------------------
__global__ void __launch_bounds__(128)
vq_prep(const float* __restrict__ cb, __bf16* __restrict__ cbh,
        float* __restrict__ norms) {
  const int n   = blockIdx.x;
  const int tid = threadIdx.x;

  f32x4 v = *(const f32x4*)(cb + (size_t)n * DDIM + tid * 4);
  u16x4 b = { f2bf(v[0]), f2bf(v[1]), f2bf(v[2]), f2bf(v[3]) };
  *(bf16x4*)(cbh + (size_t)n * DDIM + tid * 4) = __builtin_bit_cast(bf16x4, b);

  float s = v[0]*v[0] + v[1]*v[1] + v[2]*v[2] + v[3]*v[3];
  #pragma unroll
  for (int off = 16; off >= 1; off >>= 1) s += __shfl_down(s, off, 32);

  __shared__ float ps[4];
  if ((tid & 31) == 0) ps[tid >> 5] = s;
  __syncthreads();
  if (tid == 0) norms[n] = ps[0] + ps[1] + ps[2] + ps[3];
}

// ---------------------------------------------------------------------------
// Kernel 2: fused (-2 z.W^T + ||W||^2) + running argmin via bf16 WMMA.
// 256 WGs x 256 threads (8 waves). Wave w owns rows [m0+16w, m0+16w+16).
// 64 N-tiles of 16 codes; 16 WMMAs per tile over the 512-deep K dim.
// B slab staged via async memory->LDS DMA (ASYNCcnt).
// ---------------------------------------------------------------------------
__global__ void __launch_bounds__(256)
vq_argmin(const float* __restrict__ z, const __bf16* __restrict__ cbh,
          const float* __restrict__ norms, int* __restrict__ idxOut) {
  // z-tile [MTILE][LDA] bf16 (130KB) + codebook slab [16][LDB] bf16 (16.6KB)
  __shared__ __align__(16) unsigned short As[MTILE * LDA];
  __shared__ __align__(16) __bf16        Bs[16 * LDB];
  __bf16* Asb = (__bf16*)As;

  const int tid  = threadIdx.x;
  const int lane = tid & 31;
  const int wave = tid >> 5;
  const int m0   = blockIdx.x * MTILE;
  const int bidx = m0 >> 10;            // batch index (m = b*1024 + h*32 + w)
  const int hw0  = m0 & 1023;           // contiguous 128-wide h*32+w window

  // ---- Stage z tile to LDS as bf16, layout [m_local][k] ----
  const float* zb = z + (size_t)bidx * DDIM * 1024 + hw0;
  for (int it = 0; it < 64; ++it) {
    int fid = it * 256 + tid;           // 16384 float4 chunks
    int c   = fid >> 5;
    int q   = fid & 31;
    f32x4 v = *(const f32x4*)(zb + (size_t)c * 1024 + q * 4);
    int mb = q * 4;
    As[(mb + 0) * LDA + c] = f2bf(v[0]);
    As[(mb + 1) * LDA + c] = f2bf(v[1]);
    As[(mb + 2) * LDA + c] = f2bf(v[2]);
    As[(mb + 3) * LDA + c] = f2bf(v[3]);
  }

  // Per-lane WMMA operand base pointers (ISA 7.12.2 layouts):
  //   A 16x32 bf16: lane L (h=L>>4, r=L&15): row r, K in {8h..8h+7} U {16+8h..+7}
  //   B 32x16 bf16: lane L: col r, K in {8h..8h+7} U {16+8h..+7}
  const int half = lane >> 4;
  const int l15  = lane & 15;
  const __bf16* Ap = Asb + (size_t)(wave * 16 + l15) * LDA + half * 8;
  const __bf16* Bp = Bs  + (size_t)l15 * LDB + half * 8;

  // Per-thread async-DMA offsets for the B slab: 1024 x 16B chunks / tile.
  // LDS VDST operand = wave-relative LDS byte address = low 32 bits of the
  // generic pointer into the __shared__ array.
  unsigned bOffG[4], bOffL[4];
  #pragma unroll
  for (int r = 0; r < 4; ++r) {
    int cid = r * 256 + tid;
    int n   = cid >> 6;                 // code row within slab (0..15)
    int k   = (cid & 63) << 3;          // bf16 column (0..504 step 8)
    bOffG[r] = (unsigned)(n * (DDIM * 2) + k * 2);          // bytes from cbh
    bOffL[r] = (unsigned)(uintptr_t)(const void*)(Bs + (size_t)n * LDB + k);
  }

  float bestv[8];
  int   besti[8];
  #pragma unroll
  for (int v = 0; v < 8; ++v) { bestv[v] = 3.0e38f; besti[v] = 0; }

  for (int t = 0; t < 64; ++t) {
    const int n0 = t * 16;
    __syncthreads();                    // prior tile's B reads complete

    // ---- Async DMA: 16 codes x 512-deep bf16 slab, memory -> LDS ----
    const unsigned tb = (unsigned)t * 16384u;   // n0 * DDIM * 2 bytes
    #pragma unroll
    for (int r = 0; r < 4; ++r) {
      unsigned gv = bOffG[r] + tb;
      asm volatile("global_load_async_to_lds_b128 %0, %1, %2"
                   :: "v"(bOffL[r]), "v"(gv), "s"(cbh)
                   : "memory");
    }
    asm volatile("s_wait_asynccnt 0" ::: "memory");
    __syncthreads();

    f32x8 acc = {};
    #pragma unroll
    for (int kk = 0; kk < DDIM; kk += 32) {
      bf16x8 a0 = *(const bf16x8*)(Ap + kk);        // K = kk + 8h + 0..7
      bf16x8 a1 = *(const bf16x8*)(Ap + kk + 16);   // K = kk + 16 + 8h + 0..7
      bf16x8 b0 = *(const bf16x8*)(Bp + kk);        // K = kk + 8h + 0..7
      bf16x8 b1 = *(const bf16x8*)(Bp + kk + 16);   // K = kk + 16 + 8h + 0..7
      bf16x16 A16 = __builtin_shufflevector(a0, a1,
          0,1,2,3,4,5,6,7,8,9,10,11,12,13,14,15);
      bf16x16 B16 = __builtin_shufflevector(b0, b1,
          0,1,2,3,4,5,6,7,8,9,10,11,12,13,14,15);
      acc = __builtin_amdgcn_wmma_f32_16x16x32_bf16(
          false, A16, false, B16, (short)0, acc, false, false);
    }

    // D layout: lane holds N = l15, rows M = v + 8*half
    const int   n   = n0 + l15;
    const float nrm = norms[n];
    #pragma unroll
    for (int v = 0; v < 8; ++v) {
      float d = nrm - 2.0f * acc[v];
      bool better = d < bestv[v];       // strict: earlier n wins ties
      bestv[v] = better ? d : bestv[v];
      besti[v] = better ? n : besti[v];
    }
  }

  // ---- Cross-lane argmin over each 16-lane group ----
  #pragma unroll
  for (int v = 0; v < 8; ++v) {
    #pragma unroll
    for (int off = 1; off < 16; off <<= 1) {
      float ov = __shfl_xor(bestv[v], off, 32);
      int   oi = __shfl_xor(besti[v], off, 32);
      bool take = (ov < bestv[v]) || (ov == bestv[v] && oi < besti[v]);
      bestv[v] = take ? ov : bestv[v];
      besti[v] = take ? oi : besti[v];
    }
  }
  if (l15 == 0) {                       // lanes 0 (rows 0-7) and 16 (rows 8-15)
    int mg = m0 + wave * 16 + half * 8;
    #pragma unroll
    for (int v = 0; v < 8; ++v) idxOut[mg + v] = besti[v];
  }
}

// ---------------------------------------------------------------------------
// Kernel 3: gather quantized rows (f32 codebook) + emit indices as float.
// out[m*512 + d] = codebook[idx[m]][d]  (row-major, fully coalesced).
// ---------------------------------------------------------------------------
__global__ void __launch_bounds__(128)
vq_gather(const float* __restrict__ cb, const int* __restrict__ idx,
          float* __restrict__ outQ, float* __restrict__ outI) {
  const int row = blockIdx.x;
  const int tid = threadIdx.x;
  const int k   = idx[row];
  f32x4 v = *(const f32x4*)(cb + (size_t)k * DDIM + tid * 4);
  *(f32x4*)(outQ + (size_t)row * DDIM + tid * 4) = v;
  if (tid == 0) outI[row] = (float)k;
}

// ---------------------------------------------------------------------------
extern "C" void kernel_launch(void* const* d_in, const int* in_sizes, int n_in,
                              void* d_out, int out_size, void* d_ws, size_t ws_size,
                              hipStream_t stream) {
  const float* z  = (const float*)d_in[0];   // [32,512,32,32] f32
  const float* cb = (const float*)d_in[1];   // [1024,512] f32

  float* outQ = (float*)d_out;                       // 16,777,216 f32
  float* outI = outQ + (size_t)MROWS * DDIM;         // 32,768 indices as f32

  // workspace: bf16 codebook (1MB) | norms (4KB) | indices (128KB)
  __bf16* cbh   = (__bf16*)d_ws;
  float*  norms = (float*)((char*)d_ws + (size_t)KCODES * DDIM * 2);
  int*    idx   = (int*)((char*)d_ws + (size_t)KCODES * DDIM * 2 + KCODES * 4);

  vq_prep  <<<KCODES,        128, 0, stream>>>(cb, cbh, norms);
  vq_argmin<<<MROWS / MTILE, 256, 0, stream>>>(z, cbh, norms, idx);
  vq_gather<<<MROWS,         128, 0, stream>>>(cb, idx, outQ, outI);
}
